// GCNDecoder_3178275799752
// MI455X (gfx1250) — matrix operand order repaired and checked by reading, not compiled, not run on hardware
//
#include <hip/hip_runtime.h>

typedef __attribute__((ext_vector_type(16))) _Float16 v16h;
typedef __attribute__((ext_vector_type(8)))  _Float16 v8h;
typedef __attribute__((ext_vector_type(8)))  float    v8f;
typedef long long i64;

#define D_FEAT 128
#define K_TOT  256            // 2*D
#define TILE_M 16
#define APAD   8
#define ASTR   (K_TOT + APAD) // 264 halfs -> 528B row stride, conflict-free frag reads

__global__ __launch_bounds__(256) void edge_mlp_wmma(
    const float* __restrict__ x,          // [N,128]
    const i64*   __restrict__ edge_index, // [2,E]
    const float* __restrict__ W1,         // [256,128]
    const float* __restrict__ b1,         // [128]
    const float* __restrict__ W2,         // [128,1]
    const float* __restrict__ b2,         // [1]
    float*       __restrict__ out,        // [E]
    int E, int numTiles)
{
    __shared__ _Float16 A_hi[2][TILE_M][ASTR];   // double-buffered A tiles
    __shared__ _Float16 A_lo[2][TILE_M][ASTR];
    __shared__ float    out_acc[TILE_M];

    const int tid   = threadIdx.x;
    const int wv    = tid >> 5;          // wave 0..7 -> column group
    const int ln    = tid & 31;
    const int nloc  = ln & 15;           // N within wave tile / M row for A frags
    const int khalf = ln >> 4;           // which K-half this lane holds
    const int col   = (wv << 4) + nloc;  // hidden column 0..127

    // ---- one-time: B fragments (W1 columns) into registers, split f16 hi/lo.
    v16h Bhi[8], Blo[8];
    {
        const int krowbase = khalf << 4;   // 0 or 16
        #pragma unroll
        for (int k = 0; k < 8; ++k) {
            #pragma unroll
            for (int j = 0; j < 16; ++j) {
                float v = W1[(k * 32 + krowbase + j) * D_FEAT + col];
                _Float16 h = (_Float16)v;
                _Float16 l = (_Float16)(v - (float)h);
                Bhi[k][j] = h;
                Blo[k][j] = l;
            }
        }
    }
    const float bias = b1[col];
    const float w2v  = W2[col];
    const float b2v  = b2[0];

    // gather assignment: 16 threads per edge row, 16 floats per thread
    const int trow = tid >> 4;
    const int part = tid & 15;
    const int koff = khalf << 4;   // 0 or 16 halfs inside A K-block

    float4 f0, f1, f2, f3;         // in-flight gather for the *next* tile

    // load the edge index this thread needs for `tile` (clamped in-bounds)
    auto loadIndex = [&](int tile) -> i64 {
        long e  = (long)tile * TILE_M + trow;
        long ec = (e < (long)E) ? e : (long)(E - 1);
        return (part < 8) ? edge_index[ec] : edge_index[(long)E + ec];
    };
    // issue feature gather loads for a pre-loaded node index
    auto issueFeat = [&](i64 node) {
        const float4* src = (const float4*)(x + (long)node * D_FEAT + ((part & 7) << 4));
        f0 = src[0]; f1 = src[1]; f2 = src[2]; f3 = src[3];
    };
    // convert in-flight gather to split f16 and store into LDS buffer `buf`
    auto stash = [&](int buf) {
        float f[16] = { f0.x, f0.y, f0.z, f0.w, f1.x, f1.y, f1.z, f1.w,
                        f2.x, f2.y, f2.z, f2.w, f3.x, f3.y, f3.z, f3.w };
        v8h h0, h1, l0, l1;
        #pragma unroll
        for (int j = 0; j < 8; ++j) {
            _Float16 a = (_Float16)f[j];
            h0[j] = a; l0[j] = (_Float16)(f[j]     - (float)a);
            _Float16 b = (_Float16)f[j + 8];
            h1[j] = b; l1[j] = (_Float16)(f[j + 8] - (float)b);
        }
        _Float16* dh = &A_hi[buf][trow][part << 4];
        _Float16* dl = &A_lo[buf][trow][part << 4];
        *(v8h*)(dh)     = h0;  *(v8h*)(dh + 8) = h1;
        *(v8h*)(dl)     = l0;  *(v8h*)(dl + 8) = l1;
    };

    // ---- prologue: tile t0 into buffer 0; index for t0+stride in flight
    const int t0     = blockIdx.x;
    const int stride = (int)gridDim.x;
    i64 node = loadIndex(t0);
    issueFeat(node);
    stash(0);
    node = loadIndex(t0 + stride);          // one-tile-ahead index pipeline
    if (tid < TILE_M) out_acc[tid] = 0.f;
    __syncthreads();

    int buf = 0;
    for (int tile = t0; tile < numTiles; tile += stride) {
        const bool hasNext = (tile + stride) < numTiles;   // uniform across block
        if (hasNext) issueFeat(node);       // index loaded a full tile ago -> no stall
        node = loadIndex(tile + 2 * stride); // clamped, safe past the end

        // ---- 16x16 accumulator per wave, K=256 in 8 steps of 32, split-f16
        v8f c = {};
        const _Float16* Ah = &A_hi[buf][nloc][0];
        const _Float16* Al = &A_lo[buf][nloc][0];
        #pragma unroll
        for (int k = 0; k < 8; ++k) {
            const int kb = k << 5;
            v8h a0  = *(const v8h*)(Ah + kb + koff);
            v8h a1  = *(const v8h*)(Ah + kb + 8 + koff);
            v8h g0  = *(const v8h*)(Al + kb + koff);
            v8h g1  = *(const v8h*)(Al + kb + 8 + koff);
            v16h ahi = __builtin_shufflevector(a0, a1, 0,1,2,3,4,5,6,7,8,9,10,11,12,13,14,15);
            v16h alo = __builtin_shufflevector(g0, g1, 0,1,2,3,4,5,6,7,8,9,10,11,12,13,14,15);
            c = __builtin_amdgcn_wmma_f32_16x16x32_f16(false, ahi, false, Bhi[k], (short)0, c, false, false);
            c = __builtin_amdgcn_wmma_f32_16x16x32_f16(false, ahi, false, Blo[k], (short)0, c, false, false);
            c = __builtin_amdgcn_wmma_f32_16x16x32_f16(false, alo, false, Bhi[k], (short)0, c, false, false);
        }

        // ---- epilogue: +b1, ReLU, *W2[col], butterfly-reduce over 16 columns
        #pragma unroll
        for (int r = 0; r < 8; ++r) {   // row M = r + 8*khalf, N = col
            float v = c[r] + bias;
            v = v > 0.f ? v : 0.f;
            float p = v * w2v;
            p += __shfl_xor(p, 1, 32);
            p += __shfl_xor(p, 2, 32);
            p += __shfl_xor(p, 4, 32);
            p += __shfl_xor(p, 8, 32);
            if (nloc == 0) atomicAdd(&out_acc[(khalf << 3) + r], p);
        }
        __syncthreads();   // B1: out_acc complete; prev readers of A[buf^1] long done

        if (tid < TILE_M) {
            long e = (long)tile * TILE_M + tid;
            if (e < (long)E) out[e] = out_acc[tid] + b2v;
            out_acc[tid] = 0.f;
        }
        if (hasNext) stash(buf ^ 1);   // drains the overlapped feature gather
        __syncthreads();   // B2: A[buf^1] tile + zeroed out_acc visible
        buf ^= 1;
    }
}

extern "C" void kernel_launch(void* const* d_in, const int* in_sizes, int n_in,
                              void* d_out, int out_size, void* d_ws, size_t ws_size,
                              hipStream_t stream) {
    const float* x  = (const float*)d_in[0];
    const i64*   ei = (const i64*)d_in[1];
    const float* W1 = (const float*)d_in[2];
    const float* b1 = (const float*)d_in[3];
    const float* W2 = (const float*)d_in[4];
    const float* b2 = (const float*)d_in[5];
    float* out = (float*)d_out;

    int E = in_sizes[1] / 2;
    int numTiles = (E + TILE_M - 1) / TILE_M;
    int blocks = numTiles < 1600 ? numTiles : 1600;   // persistent: ~25 tiles/WG amortizes B preload
    edge_mlp_wmma<<<blocks, 256, 0, stream>>>(x, ei, W1, b1, W2, b2, out, E, numTiles);
}
